// ProgressiveST_TR_Block_36876589204197
// MI455X (gfx1250) — matrix operand order repaired and checked by reading, not compiled, not run on hardware
//
#include <hip/hip_runtime.h>
#include <hip/hip_bf16.h>

// ---------------- problem constants ----------------
constexpr int kB = 8, kT = 96, kE = 96, kC = 256, kH = 8, kD = 32, kHid = 1024;
constexpr int kN = kB * kT * kE;        // 73728 tokens
constexpr float kEps = 1e-5f;

typedef __attribute__((ext_vector_type(16))) __bf16 v16bf;
typedef __attribute__((ext_vector_type(8)))  __bf16 v8bf;
typedef __attribute__((ext_vector_type(8)))  float  v8f;
typedef int v4i_gcc __attribute__((vector_size(16)));   // matches builtin param type

// A-operand (16-bit A 16x32): lane<16 -> K {0..7,16..23}; lane>=16 -> {8..15,24..31}
// caller passes p = rowbase + 8*(lane>>4); we fetch [p,p+8) and [p+16,p+24)
__device__ __forceinline__ v16bf load_a16(const __bf16* p) {
  v8bf lo = *(const v8bf*)(p);
  v8bf hi = *(const v8bf*)(p + 16);
  v16bf r;
#pragma unroll
  for (int i = 0; i < 8; ++i) { r[i] = lo[i]; r[i + 8] = hi[i]; }
  return r;
}
// B-operand (16-bit B 32x16): lane n = L%16; elements i -> K = 8*(L>>4)+i (i<8), 16+8*(L>>4)+(i-8) (i>=8), as packed by pack_w
__device__ __forceinline__ v16bf load_b16(const __bf16* p) {
  v8bf lo = *(const v8bf*)(p);
  v8bf hi = *(const v8bf*)(p + 8);
  v16bf r;
#pragma unroll
  for (int i = 0; i < 8; ++i) { r[i] = lo[i]; r[i + 8] = hi[i]; }
  return r;
}
__device__ __forceinline__ v8f wmma_bf16(v16bf a, v16bf b, v8f c) {
  return __builtin_amdgcn_wmma_f32_16x16x32_bf16(false, a, false, b, (short)0, c, false, false);
}

// ---------------- async global->LDS staging (CDNA5 ASYNCcnt path) ----------------
#if defined(__HIP_DEVICE_COMPILE__) && __has_builtin(__builtin_amdgcn_global_load_async_to_lds_b128)
#define USE_ASYNC_LDS 1
#endif

__device__ __forceinline__ void async_copy_b128(const __bf16* gp, __bf16* lp) {
#ifdef USE_ASYNC_LDS
  __builtin_amdgcn_global_load_async_to_lds_b128((v4i_gcc*)gp, (v4i_gcc*)lp, 0, 0);
#else
  *(v8bf*)lp = *(const v8bf*)gp;
#endif
}
__device__ __forceinline__ void async_wait0() {
#ifdef USE_ASYNC_LDS
#if __has_builtin(__builtin_amdgcn_s_wait_asynccnt)
  __builtin_amdgcn_s_wait_asynccnt(0);
#else
  asm volatile("s_wait_asynccnt 0x0" ::: "memory");
#endif
#endif
}

// ---------------- weight packing: W[K,Co] f32 -> WMMA-B-layout bf16 tiles ----------------
__global__ __launch_bounds__(256) void pack_w(const float* __restrict__ W,
                                              __bf16* __restrict__ out, int K, int Co) {
  size_t idx = (size_t)blockIdx.x * 256 + threadIdx.x;
  size_t total = (size_t)K * Co;
  if (idx >= total) return;
  int i    = (int)(idx & 15);
  int lane = (int)((idx >> 4) & 31);
  int KT   = K >> 5;
  int kt   = (int)((idx >> 9) % KT);
  int nt   = (int)(idx / ((size_t)KT << 9));
  int g = lane >> 4, ln = lane & 15;
  int k = kt * 32 + (i < 8 ? 8 * g + i : 16 + 8 * g + (i - 8));
  int n = nt * 16 + ln;
  out[idx] = (__bf16)W[(size_t)k * Co + n];
}

// ---------------- block reduction over 256 threads ----------------
__device__ __forceinline__ float blk_sum256(float v, float* red) {
  int c = threadIdx.x;
  red[c] = v; __syncthreads();
#pragma unroll
  for (int s = 128; s > 0; s >>= 1) {
    if (c < s) red[c] += red[c + s];
    __syncthreads();
  }
  float r = red[0]; __syncthreads();
  return r;
}

// ---------------- LayerNorm(row of 256) -> bf16 ----------------
__global__ __launch_bounds__(256) void ln_to_bf16(const float* __restrict__ x,
                                                  const float* __restrict__ gg,
                                                  const float* __restrict__ bb,
                                                  __bf16* __restrict__ out) {
  __shared__ float red[256];
  size_t row = blockIdx.x;
  int c = threadIdx.x;
  float v = x[row * kC + c];
  float sum = blk_sum256(v, red);
  float sq  = blk_sum256(v * v, red);
  float mean = sum * (1.0f / kC);
  float var  = sq * (1.0f / kC) - mean * mean;
  out[row * kC + c] = (__bf16)((v - mean) * rsqrtf(var + kEps) * gg[c] + bb[c]);
}

// ---------------- generic WMMA GEMM: 32x64 tile per wave, 4 waves/block ----------------
constexpr int EPI_F32 = 0, EPI_BF16 = 1, EPI_GELU = 2, EPI_RES = 3, EPI_VTS = 4, EPI_VTT = 5;

template <int EPI>
__global__ __launch_bounds__(128) void gemm_wmma(const __bf16* __restrict__ A,
                                                 const __bf16* __restrict__ Bp,
                                                 const float* __restrict__ bias,
                                                 const float* __restrict__ resid,
                                                 const float* __restrict__ gamma,
                                                 void* __restrict__ outp,
                                                 int K, int Co) {
  const int lane = threadIdx.x & 31;
  const int wave = threadIdx.x >> 5;
  const int g = lane >> 4, ln = lane & 15;
  const int m0 = (blockIdx.x * 4 + wave) * 32;    // 32 rows per wave
  const int n0 = blockIdx.y * 64;                 // 64 cols per wave
  const int KT = K >> 5;
  const size_t bstride = (size_t)KT * 512;        // elements per n-tile slab
  const __bf16* arow0 = A + (size_t)(m0 + ln) * K + 8 * g;
  const __bf16* arow1 = arow0 + (size_t)16 * K;
  const __bf16* bp = Bp + ((size_t)(blockIdx.y * 4) * KT * 32 + lane) * 16;
  v8f acc[2][4] = {};
  for (int kt = 0; kt < KT; ++kt) {
    v16bf a0 = load_a16(arow0 + kt * 32);
    v16bf a1 = load_a16(arow1 + kt * 32);
    const __bf16* bk = bp + (size_t)kt * 512;
    v16bf b0 = load_b16(bk);
    v16bf b1 = load_b16(bk + bstride);
    v16bf b2 = load_b16(bk + 2 * bstride);
    v16bf b3 = load_b16(bk + 3 * bstride);
    acc[0][0] = wmma_bf16(a0, b0, acc[0][0]);
    acc[0][1] = wmma_bf16(a0, b1, acc[0][1]);
    acc[0][2] = wmma_bf16(a0, b2, acc[0][2]);
    acc[0][3] = wmma_bf16(a0, b3, acc[0][3]);
    acc[1][0] = wmma_bf16(a1, b0, acc[1][0]);
    acc[1][1] = wmma_bf16(a1, b1, acc[1][1]);
    acc[1][2] = wmma_bf16(a1, b2, acc[1][2]);
    acc[1][3] = wmma_bf16(a1, b3, acc[1][3]);
  }
  float gm = (EPI == EPI_RES) ? (gamma ? gamma[0] : 1.0f) : 0.0f;
#pragma unroll
  for (int mt = 0; mt < 2; ++mt) {
#pragma unroll
    for (int nt = 0; nt < 4; ++nt) {
      int col = n0 + nt * 16 + ln;
      float bval = bias ? bias[col] : 0.0f;
#pragma unroll
      for (int r = 0; r < 8; ++r) {
        int row = m0 + mt * 16 + r + 8 * g;
        size_t idx = (size_t)row * Co + col;
        float v = acc[mt][nt][r] + bval;
        if (EPI == EPI_F32) {
          ((float*)outp)[idx] = v;
        } else if (EPI == EPI_BF16) {
          ((__bf16*)outp)[idx] = (__bf16)v;
        } else if (EPI == EPI_GELU) {
          ((__bf16*)outp)[idx] = (__bf16)(0.5f * v * (1.0f + erff(v * 0.70710678118654752f)));
        } else if (EPI == EPI_RES) {
          ((float*)outp)[idx] = resid[idx] + gm * v;
        } else if (EPI == EPI_VTS) {  // V^T for spatial attn: [b,t,h][d][e]
          int e = row % kE; int tf = (row / kE) % kT; int b_ = row / (kE * kT);
          int h = col / kD, d = col % kD;
          ((__bf16*)outp)[((((size_t)(b_ * kT + tf)) * kH + h) * kD + d) * kE + e] = (__bf16)v;
        } else if (EPI == EPI_VTT) {  // V^T for temporal attn: [b,e,h][d][t]
          int e = row % kE; int tf = (row / kE) % kT; int b_ = row / (kE * kT);
          int h = col / kD, d = col % kD;
          ((__bf16*)outp)[((((size_t)(b_ * kE + e)) * kH + h) * kD + d) * kT + tf] = (__bf16)v;
        }
      }
    }
  }
}

// ---------------- spatial masked-cosine attention, one block per (b,t,h) ----------------
__global__ __launch_bounds__(256) void attn_spatial(const __bf16* __restrict__ q,
                                                    const __bf16* __restrict__ k,
                                                    const __bf16* __restrict__ vt,
                                                    const float* __restrict__ adj,
                                                    __bf16* __restrict__ o) {
  __shared__ __align__(16) __bf16 sQ[96 * 32];
  __shared__ __align__(16) __bf16 sK[96 * 32];
  __shared__ __align__(16) __bf16 sV[32 * 96];
  __shared__ __align__(16) __bf16 sAttn[96 * 104];
  int bid = blockIdx.x;
  int h = bid % kH;
  int tf = (bid / kH) % kT;
  int b = bid / (kH * kT);
  int tid = threadIdx.x;
  int wave = tid >> 5, lane = tid & 31;
  int g = lane >> 4, ln = lane & 15;
  size_t tokbase = (size_t)(b * kT + tf) * kE;
  const __bf16* qb = q + tokbase * kC + h * kD;
  const __bf16* kb = k + tokbase * kC + h * kD;
  const __bf16* vb = vt + ((size_t)(b * kT + tf) * kH + h) * (kD * kE);

  // async-stage Q,K (96 rows x 32 elems, 16B chunks)
  for (int c = tid; c < 384; c += 256) {
    int r = c >> 2, ch = (c & 3) * 8;
    async_copy_b128(qb + (size_t)r * kC + ch, &sQ[r * 32 + ch]);
    async_copy_b128(kb + (size_t)r * kC + ch, &sK[r * 32 + ch]);
  }
  async_wait0();
  __syncthreads();
  // kick off V^T staging; overlaps with score WMMAs
  for (int c = tid; c < 384; c += 256) {
    int r = c / 12, ch = (c % 12) * 8;
    async_copy_b128(vb + (size_t)r * kE + ch, &sV[r * kE + ch]);
  }

  for (int tile = wave; tile < 36; tile += 8) {   // 6x6 score tiles, K=D=32
    int qt = tile / 6, kt = tile % 6;
    v16bf a  = load_a16(&sQ[(qt * 16 + ln) * 32 + 8 * g]);
    v16bf bm = load_a16(&sK[(kt * 16 + ln) * 32 + 8 * g]);
    v8f acc = {};
    acc = wmma_bf16(a, bm, acc);
#pragma unroll
    for (int r = 0; r < 8; ++r) {
      int qr = qt * 16 + r + 8 * g;
      int kc = kt * 16 + ln;
      float s = (acc[r] * adj[qr * kE + kc] + 1.0f) * 0.5f;
      sAttn[qr * 104 + kc] = (__bf16)s;
    }
  }
  async_wait0();
  __syncthreads();

  for (int tile = wave; tile < 12; tile += 8) {   // O = attn @ V : 6x2 tiles, 3 k-steps
    int qt = tile >> 1, dt = tile & 1;
    v8f acc = {};
#pragma unroll
    for (int ks = 0; ks < 3; ++ks) {
      v16bf a  = load_a16(&sAttn[(qt * 16 + ln) * 104 + ks * 32 + 8 * g]);
      v16bf bm = load_a16(&sV[(dt * 16 + ln) * kE + ks * 32 + 8 * g]);
      acc = wmma_bf16(a, bm, acc);
    }
#pragma unroll
    for (int r = 0; r < 8; ++r) {
      int qr = qt * 16 + r + 8 * g;
      o[(tokbase + qr) * kC + h * kD + dt * 16 + ln] = (__bf16)acc[r];
    }
  }
}

// ---------------- temporal softmax attention, one block per (b,e,h) ----------------
__global__ __launch_bounds__(256) void attn_temporal(const __bf16* __restrict__ q,
                                                     const __bf16* __restrict__ k,
                                                     const __bf16* __restrict__ vt,
                                                     __bf16* __restrict__ o) {
  __shared__ __align__(16) __bf16 sQ[96 * 32];
  __shared__ __align__(16) __bf16 sK[96 * 32];
  __shared__ __align__(16) __bf16 sV[32 * 96];
  __shared__ __align__(16) __bf16 sAttn[96 * 104];
  int bid = blockIdx.x;
  int h = bid % kH;
  int e = (bid / kH) % kE;
  int b = bid / (kH * kE);
  int tid = threadIdx.x;
  int wave = tid >> 5, lane = tid & 31;
  int g = lane >> 4, ln = lane & 15;
  const size_t rstride = (size_t)kE * kC;         // stride between time rows
  const __bf16* qb = q + ((size_t)b * kT * kE + e) * kC + h * kD;
  const __bf16* kb = k + ((size_t)b * kT * kE + e) * kC + h * kD;
  const __bf16* vb = vt + ((size_t)(b * kE + e) * kH + h) * (kD * kT);

  for (int c = tid; c < 384; c += 256) {
    int r = c >> 2, ch = (c & 3) * 8;
    async_copy_b128(qb + (size_t)r * rstride + ch, &sQ[r * 32 + ch]);
    async_copy_b128(kb + (size_t)r * rstride + ch, &sK[r * 32 + ch]);
  }
  async_wait0();
  __syncthreads();
  for (int c = tid; c < 384; c += 256) {
    int r = c / 12, ch = (c % 12) * 8;
    async_copy_b128(vb + (size_t)r * kT + ch, &sV[r * kT + ch]);
  }

  for (int tile = wave; tile < 36; tile += 8) {   // raw scores (bf16 in LDS)
    int qt = tile / 6, kt = tile % 6;
    v16bf a  = load_a16(&sQ[(qt * 16 + ln) * 32 + 8 * g]);
    v16bf bm = load_a16(&sK[(kt * 16 + ln) * 32 + 8 * g]);
    v8f acc = {};
    acc = wmma_bf16(a, bm, acc);
#pragma unroll
    for (int r = 0; r < 8; ++r) {
      int qr = qt * 16 + r + 8 * g;
      sAttn[qr * 104 + kt * 16 + ln] = (__bf16)(acc[r] * 0.17677669529663687f);  // d^-0.5
    }
  }
  __syncthreads();
  if (tid < 96) {                                 // row softmax, fp32 accumulation
    int row = tid;
    float mx = -1e30f;
    for (int j = 0; j < 96; ++j) mx = fmaxf(mx, (float)sAttn[row * 104 + j]);
    float sum = 0.0f;
    for (int j = 0; j < 96; ++j) sum += expf((float)sAttn[row * 104 + j] - mx);
    float inv = 1.0f / sum;
    for (int j = 0; j < 96; ++j)
      sAttn[row * 104 + j] = (__bf16)(expf((float)sAttn[row * 104 + j] - mx) * inv);
  }
  async_wait0();
  __syncthreads();

  for (int tile = wave; tile < 12; tile += 8) {
    int qt = tile >> 1, dt = tile & 1;
    v8f acc = {};
#pragma unroll
    for (int ks = 0; ks < 3; ++ks) {
      v16bf a  = load_a16(&sAttn[(qt * 16 + ln) * 104 + ks * 32 + 8 * g]);
      v16bf bm = load_a16(&sV[(dt * 16 + ln) * kT + ks * 32 + 8 * g]);
      acc = wmma_bf16(a, bm, acc);
    }
#pragma unroll
    for (int r = 0; r < 8; ++r) {
      int qr = qt * 16 + r + 8 * g;
      o[((size_t)(b * kT + qr) * kE + e) * kC + h * kD + dt * 16 + ln] = (__bf16)acc[r];
    }
  }
}

// ---- x1 = x + LN(x,n1) + s_gamma * LN(p, s_ng, s_nb), p = o@s_wp+s_bp ----
__global__ __launch_bounds__(256) void spatial_residual(const float* __restrict__ x,
    const float* __restrict__ p, const float* __restrict__ n1g, const float* __restrict__ n1b,
    const float* __restrict__ sng, const float* __restrict__ snb,
    const float* __restrict__ sgamma, float* __restrict__ out) {
  __shared__ float red[256];
  size_t row = blockIdx.x;
  int c = threadIdx.x;
  float xv = x[row * kC + c];
  float pv = p[row * kC + c];
  float sx  = blk_sum256(xv, red);
  float sxx = blk_sum256(xv * xv, red);
  float sp  = blk_sum256(pv, red);
  float spp = blk_sum256(pv * pv, red);
  float mx = sx * (1.0f / kC), vx = sxx * (1.0f / kC) - mx * mx;
  float mp = sp * (1.0f / kC), vp = spp * (1.0f / kC) - mp * mp;
  float xs  = (xv - mx) * rsqrtf(vx + kEps) * n1g[c] + n1b[c];
  float lnp = (pv - mp) * rsqrtf(vp + kEps) * sng[c] + snb[c];
  out[row * kC + c] = xv + xs + sgamma[0] * lnp;
}

// ---- linear T-interp 96->48 (align_corners=False) == pairwise mean ----
__global__ __launch_bounds__(256) void downsample_t(const float* __restrict__ x,
                                                    float* __restrict__ out) {
  size_t idx = (size_t)blockIdx.x * 256 + threadIdx.x;
  const size_t total = (size_t)kB * 48 * kE * kC;
  if (idx >= total) return;
  int c = (int)(idx % kC);
  size_t r = idx / kC;
  int e = (int)(r % kE);
  size_t r2 = r / kE;
  int ti = (int)(r2 % 48);
  int b = (int)(r2 / 48);
  size_t base = (((size_t)b * kT + 2 * ti) * kE + e) * kC + c;
  out[idx] = 0.5f * (x[base] + x[base + (size_t)kE * kC]);
}

// ---------------- launcher ----------------
extern "C" void kernel_launch(void* const* d_in, const int* in_sizes, int n_in,
                              void* d_out, int out_size, void* d_ws, size_t ws_size,
                              hipStream_t stream) {
  const float* x       = (const float*)d_in[0];
  const float* adj     = (const float*)d_in[1];
  const float* n1g     = (const float*)d_in[2];
  const float* n1b     = (const float*)d_in[3];
  const float* s_wq    = (const float*)d_in[4];
  const float* s_wk    = (const float*)d_in[5];
  const float* s_wv    = (const float*)d_in[6];
  const float* s_wp    = (const float*)d_in[7];
  const float* s_bp    = (const float*)d_in[8];
  const float* s_ng    = (const float*)d_in[9];
  const float* s_nb    = (const float*)d_in[10];
  const float* s_gamma = (const float*)d_in[11];
  const float* n2g     = (const float*)d_in[12];
  const float* n2b     = (const float*)d_in[13];
  const float* t_wq    = (const float*)d_in[14];
  const float* t_wk    = (const float*)d_in[15];
  const float* t_wv    = (const float*)d_in[16];
  const float* t_wp    = (const float*)d_in[17];
  const float* t_bp    = (const float*)d_in[18];
  const float* t_gamma = (const float*)d_in[19];
  const float* n3g     = (const float*)d_in[20];
  const float* n3b     = (const float*)d_in[21];
  const float* m_w1    = (const float*)d_in[22];
  const float* m_b1    = (const float*)d_in[23];
  const float* m_w2    = (const float*)d_in[24];
  const float* m_b2    = (const float*)d_in[25];
  (void)in_sizes; (void)n_in; (void)out_size; (void)ws_size;

  char* ws = (char*)d_ws;
  const size_t NC = (size_t)kN * kC;
  __bf16* pw  = (__bf16*)ws;
  size_t off = (size_t)2 * 1048576;              // packed weights: 1,048,576 bf16
  __bf16* bf1 = (__bf16*)(ws + off); off += NC * 2;   // LN output (xs/xt/xm)
  __bf16* bf2 = (__bf16*)(ws + off); off += NC * 2;   // Q
  __bf16* bf3 = (__bf16*)(ws + off); off += NC * 2;   // K
  __bf16* bf4 = (__bf16*)(ws + off); off += NC * 2;   // V^T
  __bf16* bf5 = (__bf16*)(ws + off); off += NC * 2;   // attn out
  float*  f1  = (float*)(ws + off);  off += NC * 4;   // running residual
  float*  f2  = (float*)(ws + off);  off += NC * 4;   // proj / residual scratch
  __bf16* hbuf = (__bf16*)(ws + off);                 // MLP hidden [N, 1024] bf16

  __bf16* p_swq = pw + 0 * 65536;
  __bf16* p_swk = pw + 1 * 65536;
  __bf16* p_swv = pw + 2 * 65536;
  __bf16* p_swp = pw + 3 * 65536;
  __bf16* p_twq = pw + 4 * 65536;
  __bf16* p_twk = pw + 5 * 65536;
  __bf16* p_twv = pw + 6 * 65536;
  __bf16* p_twp = pw + 7 * 65536;
  __bf16* p_mw1 = pw + 8 * 65536;
  __bf16* p_mw2 = pw + 8 * 65536 + 262144;

  pack_w<<<256, 256, 0, stream>>>(s_wq, p_swq, kC, kC);
  pack_w<<<256, 256, 0, stream>>>(s_wk, p_swk, kC, kC);
  pack_w<<<256, 256, 0, stream>>>(s_wv, p_swv, kC, kC);
  pack_w<<<256, 256, 0, stream>>>(s_wp, p_swp, kC, kC);
  pack_w<<<256, 256, 0, stream>>>(t_wq, p_twq, kC, kC);
  pack_w<<<256, 256, 0, stream>>>(t_wk, p_twk, kC, kC);
  pack_w<<<256, 256, 0, stream>>>(t_wv, p_twv, kC, kC);
  pack_w<<<256, 256, 0, stream>>>(t_wp, p_twp, kC, kC);
  pack_w<<<1024, 256, 0, stream>>>(m_w1, p_mw1, kC, kHid);
  pack_w<<<1024, 256, 0, stream>>>(m_w2, p_mw2, kHid, kC);

  const int MB = kN / 128;                        // 576: 4 waves x 32 rows per block
  dim3 gemmC(MB, kC / 64), gemmH(MB, kHid / 64), b128(128);

  // ---- spatial attention block ----
  ln_to_bf16<<<kN, 256, 0, stream>>>(x, n1g, n1b, bf1);
  gemm_wmma<EPI_BF16><<<gemmC, b128, 0, stream>>>(bf1, p_swq, nullptr, nullptr, nullptr, bf2, kC, kC);
  gemm_wmma<EPI_BF16><<<gemmC, b128, 0, stream>>>(bf1, p_swk, nullptr, nullptr, nullptr, bf3, kC, kC);
  gemm_wmma<EPI_VTS ><<<gemmC, b128, 0, stream>>>(bf1, p_swv, nullptr, nullptr, nullptr, bf4, kC, kC);
  attn_spatial<<<kB * kT * kH, 256, 0, stream>>>(bf2, bf3, bf4, adj, bf5);
  gemm_wmma<EPI_F32 ><<<gemmC, b128, 0, stream>>>(bf5, p_swp, s_bp, nullptr, nullptr, f2, kC, kC);
  spatial_residual<<<kN, 256, 0, stream>>>(x, f2, n1g, n1b, s_ng, s_nb, s_gamma, f1);

  // ---- temporal attention block ----
  ln_to_bf16<<<kN, 256, 0, stream>>>(f1, n2g, n2b, bf1);
  gemm_wmma<EPI_BF16><<<gemmC, b128, 0, stream>>>(bf1, p_twq, nullptr, nullptr, nullptr, bf2, kC, kC);
  gemm_wmma<EPI_BF16><<<gemmC, b128, 0, stream>>>(bf1, p_twk, nullptr, nullptr, nullptr, bf3, kC, kC);
  gemm_wmma<EPI_VTT ><<<gemmC, b128, 0, stream>>>(bf1, p_twv, nullptr, nullptr, nullptr, bf4, kC, kC);
  attn_temporal<<<kB * kE * kH, 256, 0, stream>>>(bf2, bf3, bf4, bf5);
  gemm_wmma<EPI_RES ><<<gemmC, b128, 0, stream>>>(bf5, p_twp, t_bp, f1, t_gamma, f2, kC, kC);

  // ---- MLP ----
  ln_to_bf16<<<kN, 256, 0, stream>>>(f2, n3g, n3b, bf1);
  gemm_wmma<EPI_GELU><<<gemmH, b128, 0, stream>>>(bf1, p_mw1, m_b1, nullptr, nullptr, hbuf, kC, kHid);
  gemm_wmma<EPI_RES ><<<gemmC, b128, 0, stream>>>(hbuf, p_mw2, m_b2, f2, nullptr, f1, kHid, kC);

  // ---- temporal downsample ----
  const size_t outN = (size_t)kB * 48 * kE * kC;
  downsample_t<<<(unsigned)((outN + 255) / 256), 256, 0, stream>>>(f1, (float*)d_out);
}
